// MultiHeadAttention_68126771249166
// MI455X (gfx1250) — matrix operand order repaired and checked by reading, not compiled, not run on hardware
//
#include <hip/hip_runtime.h>
#include <hip/hip_bf16.h>
#include <cstdint>
#include <cstddef>

// Problem constants (fixed by the reference): B=4, T=2048, C=1024, H=16, hd=64
#define BB 4
#define TT 2048
#define CC 1024
#define HH 16
#define HD 64
#define MROWS (BB * TT)   // 8192

typedef __attribute__((ext_vector_type(16))) __bf16 v16bf;
typedef __attribute__((ext_vector_type(8)))  float  v8f;

union FragBf {
    v16bf bf;
    uint4 u4[2];
};

__device__ __forceinline__ unsigned short f2bf(float f) {
    unsigned int u = __float_as_uint(f);
    unsigned int r = u + 0x7FFFu + ((u >> 16) & 1u);  // round-to-nearest-even
    return (unsigned short)(r >> 16);
}

__device__ __forceinline__ v8f wmma_bf16(const v16bf& a, const v16bf& b, v8f c) {
    // (neg_a, A, neg_b, B, c_mod, C, reuse_a, reuse_b)
    return __builtin_amdgcn_wmma_f32_16x16x32_bf16(false, a, false, b, (short)0, c,
                                                   false, false);
}

// ---------------------------------------------------------------------------
// GEMM: Out[m, n] = sum_k A[m, k] * W[n, k]   (A: MROWSxCC, W: CCxCC row-major)
// Block: 256 threads (8 waves), tile 128(M) x 128(N), K-step 32, ping-pong LDS.
// Wave (wm 0..3, wn 0..1) owns a 32x64 sub-tile -> 8 WMMA accumulators.
// Pipeline per step: issue next global loads -> WMMAs from LDS[cur] ->
//                    store staged regs to LDS[cur^1] -> barrier.
// ---------------------------------------------------------------------------
template <bool A_BF16, bool OUT_F32, bool V_TRANS>
__global__ __launch_bounds__(256) void gemm_wmma_kernel(
    const void* __restrict__ Ap, const float* __restrict__ W,
    const float* __restrict__ bias, void* __restrict__ Outp) {
    constexpr int BM = 128, BN = 128, BK = 32, PAD = 8;
    constexpr int NSTEP = CC / BK;  // 32
    __shared__ __align__(16) unsigned short As[2][BM][BK + PAD];
    __shared__ __align__(16) unsigned short Ws[2][BN][BK + PAD];

    const int m0 = blockIdx.x * BM;
    const int n0 = blockIdx.y * BN;
    const int tid = threadIdx.x;
    const int wave = tid >> 5;
    const int lane = tid & 31;
    const int sub = lane >> 4;   // lane half (0/1)
    const int l16 = lane & 15;
    const int wm = wave & 3;     // M offset wm*32
    const int wn = wave >> 2;    // N offset wn*64

    v8f acc[2][4] = {};

    float4 aF[4];
    uint4  aH[2];
    float4 wF[4];

    auto gload = [&](int kk) {
        if constexpr (A_BF16) {
            const unsigned short* A = (const unsigned short*)Ap;
#pragma unroll
            for (int i = 0; i < 2; ++i) {
                int idx = tid + i * 256;            // 512 x (8 ushorts)
                int r = idx >> 2, c8 = (idx & 3) << 3;
                aH[i] = *(const uint4*)(A + (size_t)(m0 + r) * CC + kk + c8);
            }
        } else {
            const float* A = (const float*)Ap;
#pragma unroll
            for (int i = 0; i < 4; ++i) {
                int idx = tid + i * 256;            // 1024 x float4
                int r = idx >> 3, c4 = (idx & 7) << 2;
                aF[i] = *(const float4*)(A + (size_t)(m0 + r) * CC + kk + c4);
            }
        }
#pragma unroll
        for (int i = 0; i < 4; ++i) {               // W tile: 128x32 = 1024 float4
            int idx = tid + i * 256;
            int r = idx >> 3, c4 = (idx & 7) << 2;
            wF[i] = *(const float4*)(W + (size_t)(n0 + r) * CC + kk + c4);
        }
    };
    auto lstore = [&](int buf) {
        if constexpr (A_BF16) {
#pragma unroll
            for (int i = 0; i < 2; ++i) {
                int idx = tid + i * 256;
                int r = idx >> 2, c8 = (idx & 3) << 3;
                *(uint4*)&As[buf][r][c8] = aH[i];
            }
        } else {
#pragma unroll
            for (int i = 0; i < 4; ++i) {
                int idx = tid + i * 256;
                int r = idx >> 3, c4 = (idx & 7) << 2;
                uint2 pk;
                pk.x = (unsigned int)f2bf(aF[i].x) | ((unsigned int)f2bf(aF[i].y) << 16);
                pk.y = (unsigned int)f2bf(aF[i].z) | ((unsigned int)f2bf(aF[i].w) << 16);
                *(uint2*)&As[buf][r][c4] = pk;
            }
        }
#pragma unroll
        for (int i = 0; i < 4; ++i) {
            int idx = tid + i * 256;
            int r = idx >> 3, c4 = (idx & 7) << 2;
            uint2 pk;
            pk.x = (unsigned int)f2bf(wF[i].x) | ((unsigned int)f2bf(wF[i].y) << 16);
            pk.y = (unsigned int)f2bf(wF[i].z) | ((unsigned int)f2bf(wF[i].w) << 16);
            *(uint2*)&Ws[buf][r][c4] = pk;
        }
    };

    // prologue: tile 0 into buffer 0
    gload(0);
    lstore(0);
    __syncthreads();

    const int klo = sub ? 8 : 0;   // A-layout: lanes0-15 K{0..7,16..23}, lanes16-31 K{8..15,24..31}
    const int koff = sub * 8;      // B-layout: lanes0-15 K{0..7,16..23}, lanes16-31 K{8..15,24..31}

    for (int s = 0; s < NSTEP; ++s) {
        const int cur = s & 1;
        const bool more = (s + 1) < NSTEP;
        if (more) gload((s + 1) * BK);   // next tile's global loads in flight

        FragBf a[2], b[4];
#pragma unroll
        for (int mt = 0; mt < 2; ++mt) {
            int row = wm * 32 + mt * 16 + l16;
            a[mt].u4[0] = *(const uint4*)&As[cur][row][klo];
            a[mt].u4[1] = *(const uint4*)&As[cur][row][klo + 16];
        }
#pragma unroll
        for (int nt = 0; nt < 4; ++nt) {
            int wrow = wn * 64 + nt * 16 + l16;
            b[nt].u4[0] = *(const uint4*)&Ws[cur][wrow][koff];
            b[nt].u4[1] = *(const uint4*)&Ws[cur][wrow][koff + 16];
        }
#pragma unroll
        for (int mt = 0; mt < 2; ++mt)
#pragma unroll
            for (int nt = 0; nt < 4; ++nt)
                acc[mt][nt] = wmma_bf16(a[mt].bf, b[nt].bf, acc[mt][nt]);

        if (more) lstore(cur ^ 1);       // waits on global loads AFTER the WMMAs
        __syncthreads();
    }

    // ---- store ----
#pragma unroll
    for (int mt = 0; mt < 2; ++mt) {
#pragma unroll
        for (int nt = 0; nt < 4; ++nt) {
#pragma unroll
            for (int r = 0; r < 8; ++r) {
                int row = m0 + wm * 32 + mt * 16 + r + sub * 8;   // token index
                int col = n0 + wn * 64 + nt * 16 + l16;           // feature index
                float v = acc[mt][nt][r];
                if constexpr (OUT_F32) {
                    ((float*)Outp)[(size_t)row * CC + col] = v + bias[col];
                } else {
                    int bIdx = row >> 11;          // row / T
                    int t = row & (TT - 1);
                    int h = col >> 6;              // col / HD
                    int d = col & (HD - 1);
                    size_t addr;
                    if constexpr (V_TRANS)
                        addr = ((size_t)(bIdx * HH + h) * HD + d) * TT + t;
                    else
                        addr = ((size_t)(bIdx * HH + h) * TT + t) * HD + d;
                    ((unsigned short*)Outp)[addr] = f2bf(v);
                }
            }
        }
    }
}

// ---------------------------------------------------------------------------
// Flash attention, causal, computed as S^T = K @ Q^T. One wave per block.
// grid = (T/16 q-tiles, B*H). Q/K: [bh][T][HD] bf16, Vt: [bh][HD][T] bf16.
//
// Key trick: in the S^T C-tile, a softmax row (fixed q, keys varying) lives
// in lane q across the 8 accumulator VGPRs with key pattern
//   sub=0: {0..7,16..23}, sub=1: {8..15,24..31}
// which is EXACTLY the 16-bit WMMA A-fragment K-pattern. So exp'd P values
// pack directly into the P@V A-fragment in registers: no LDS, no barriers,
// and softmax reductions are 15 in-lane ops + one shfl_xor(16).
// ---------------------------------------------------------------------------
__global__ __launch_bounds__(32) void attn_wmma_kernel(
    const unsigned short* __restrict__ Q, const unsigned short* __restrict__ K,
    const unsigned short* __restrict__ Vt, unsigned short* __restrict__ Y) {
    const int lane = threadIdx.x;
    const int sub = lane >> 4;
    const int l16 = lane & 15;
    const int qb = blockIdx.x * 16;
    const int bh = blockIdx.y;
    const int bIdx = bh >> 4;
    const int h = bh & (HH - 1);

    const unsigned short* Qb = Q + (size_t)bh * TT * HD;
    const unsigned short* Kb = K + (size_t)bh * TT * HD;
    const unsigned short* Vb = Vt + (size_t)bh * HD * TT;

    const int klo = sub ? 8 : 0;

    // Q as B-fragments: column q = l16, K-dim rows = d (lane half 0: d{0..7,16..23}, half 1: d{8..15,24..31})
    FragBf qf[2];
#pragma unroll
    for (int dc = 0; dc < 2; ++dc) {
        const unsigned short* p = Qb + (size_t)(qb + l16) * HD + dc * 32 + klo;
        qf[dc].u4[0] = *(const uint4*)p;
        qf[dc].u4[1] = *(const uint4*)(p + 16);
    }

    // K as A-fragments: lane = key row, A-layout d-chunk pattern
    auto loadK = [&](int k0, FragBf ka[2][2]) {
#pragma unroll
        for (int t = 0; t < 2; ++t)
#pragma unroll
            for (int dc = 0; dc < 2; ++dc) {
                const unsigned short* p =
                    Kb + (size_t)(k0 + t * 16 + l16) * HD + dc * 32 + klo;
                ka[t][dc].u4[0] = *(const uint4*)p;
                ka[t][dc].u4[1] = *(const uint4*)(p + 16);
            }
    };

    float m_c = -3.0e38f, l_c = 0.0f;   // per-lane (column q) running max / sum
    v8f o[4] = {};

    const float scale = 0.125f;  // 1/sqrt(64)
    const int kend = qb + 16;    // keys 0 .. qb+15 (causal)
    const int q = qb + l16;

    FragBf ka[2][2];
    loadK(0, ka);

    for (int k0 = 0; k0 < kend; k0 += 32) {
        // ---- S^T tiles: st[t] = K[k0+t*16 .. +15] @ Q^T ----
        v8f st0 = {}, st1 = {};
        st0 = wmma_bf16(ka[0][0].bf, qf[0].bf, st0);
        st0 = wmma_bf16(ka[0][1].bf, qf[1].bf, st0);
        st1 = wmma_bf16(ka[1][0].bf, qf[0].bf, st1);
        st1 = wmma_bf16(ka[1][1].bf, qf[1].bf, st1);

        // next K tile + current V tiles in flight behind the softmax VALU
        const bool more = (k0 + 32) < kend;
        FragBf kan[2][2];
        if (more) loadK(k0 + 32, kan);
        FragBf vb[4];
#pragma unroll
        for (int dt = 0; dt < 4; ++dt) {
            const unsigned short* p =
                Vb + (size_t)(dt * 16 + l16) * TT + k0 + klo;
            vb[dt].u4[0] = *(const uint4*)p;
            vb[dt].u4[1] = *(const uint4*)(p + 16);
        }
        if (more) __builtin_prefetch(Vb + (size_t)l16 * TT + k0 + 32, 0, 1);

        // ---- per-lane softmax over this lane's 16 keys of column q ----
        // x[i]: i<8 -> key k0 + klo + i (tile 0); i>=8 -> key k0 + 16 + klo + (i-8)
        float x[16];
#pragma unroll
        for (int r = 0; r < 8; ++r) {
            int key0 = k0 + klo + r;
            int key1 = key0 + 16;
            float v0 = st0[r] * scale;
            float v1 = st1[r] * scale;
            x[r]     = (key0 > q) ? -1.0e30f : v0;
            x[r + 8] = (key1 > q) ? -1.0e30f : v1;
        }
        float mx = x[0];
#pragma unroll
        for (int i = 1; i < 16; ++i) mx = fmaxf(mx, x[i]);
        mx = fmaxf(mx, __shfl_xor(mx, 16));        // combine the two lane halves
        float m_new = fmaxf(m_c, mx);
        float alpha = __expf(m_c - m_new);
        m_c = m_new;

        float p[16], ps = 0.0f;
#pragma unroll
        for (int i = 0; i < 16; ++i) { p[i] = __expf(x[i] - m_new); ps += p[i]; }
        ps += __shfl_xor(ps, 16);
        l_c = l_c * alpha + ps;

        // ---- pack P directly as the A-fragment (layout already matches) ----
        FragBf pa;
        unsigned int pw[8];
#pragma unroll
        for (int j = 0; j < 8; ++j)
            pw[j] = (unsigned int)f2bf(p[2 * j]) | ((unsigned int)f2bf(p[2 * j + 1]) << 16);
        pa.u4[0].x = pw[0]; pa.u4[0].y = pw[1]; pa.u4[0].z = pw[2]; pa.u4[0].w = pw[3];
        pa.u4[1].x = pw[4]; pa.u4[1].y = pw[5]; pa.u4[1].z = pw[6]; pa.u4[1].w = pw[7];

        // ---- rescale O rows by alpha (broadcast alpha[row q] from lane q) ----
#pragma unroll
        for (int r = 0; r < 8; ++r) {
            float ar = __shfl(alpha, r + sub * 8);
#pragma unroll
            for (int dt = 0; dt < 4; ++dt) o[dt][r] *= ar;
        }

        // ---- O += P @ V ----
#pragma unroll
        for (int dt = 0; dt < 4; ++dt)
            o[dt] = wmma_bf16(pa.bf, vb[dt].bf, o[dt]);

        if (more) {
#pragma unroll
            for (int t = 0; t < 2; ++t)
#pragma unroll
                for (int dc = 0; dc < 2; ++dc) ka[t][dc] = kan[t][dc];
        }
    }

    // ---- normalize + store y (B,T,C) bf16 ----
#pragma unroll
    for (int r = 0; r < 8; ++r) {
        float lr = __shfl(l_c, r + sub * 8);
        float inv = 1.0f / lr;
        int t = qb + r + sub * 8;
        size_t base = ((size_t)(bIdx * TT + t)) * CC + h * HD;
#pragma unroll
        for (int dt = 0; dt < 4; ++dt)
            Y[base + dt * 16 + l16] = f2bf(o[dt][r] * inv);
    }
}

// ---------------------------------------------------------------------------
extern "C" void kernel_launch(void* const* d_in, const int* in_sizes, int n_in,
                              void* d_out, int out_size, void* d_ws, size_t ws_size,
                              hipStream_t stream) {
    (void)in_sizes; (void)n_in; (void)out_size; (void)ws_size;
    const float* xq = (const float*)d_in[0];
    const float* xk = (const float*)d_in[1];
    const float* xv = (const float*)d_in[2];
    const float* Wq = (const float*)d_in[3];
    const float* Wk = (const float*)d_in[4];
    const float* Wv = (const float*)d_in[5];
    const float* Wo = (const float*)d_in[6];
    const float* bo = (const float*)d_in[7];
    float* out = (float*)d_out;

    const size_t perBuf = (size_t)BB * HH * TT * HD;  // 8,388,608 elements
    unsigned short* Qws = (unsigned short*)d_ws;
    unsigned short* Kws = Qws + perBuf;
    unsigned short* Vtws = Kws + perBuf;
    unsigned short* Yws = Vtws + perBuf;              // (B,T,C) bf16, same size

    dim3 gGemm(MROWS / 128, CC / 128);
    dim3 bGemm(256);
    // Q, K projections -> [bh][t][d]; V projection -> [bh][d][t] (transposed)
    gemm_wmma_kernel<false, false, false><<<gGemm, bGemm, 0, stream>>>(xq, Wq, nullptr, Qws);
    gemm_wmma_kernel<false, false, false><<<gGemm, bGemm, 0, stream>>>(xk, Wk, nullptr, Kws);
    gemm_wmma_kernel<false, false, true ><<<gGemm, bGemm, 0, stream>>>(xv, Wv, nullptr, Vtws);

    dim3 gAttn(TT / 16, BB * HH);
    attn_wmma_kernel<<<gAttn, dim3(32), 0, stream>>>(Qws, Kws, Vtws, Yws);

    // Output projection: y @ Wo^T + bo -> f32
    gemm_wmma_kernel<true, true, false><<<gGemm, bGemm, 0, stream>>>(Yws, Wo, bo, out);
}
